// EXAMPLE_2_2310692405591
// MI455X (gfx1250) — matrix operand order repaired and checked by reading, not compiled, not run on hardware
//
#include <hip/hip_runtime.h>

// ---------------------------------------------------------------------------
// Spiking CNN (sinabs IAF, membrane-subtract) for MI455X / gfx1250.
//
//   K1  conv1+iaf fused (VALU, 10890 neurons x T loop) -> c1 spikes f16
//   K2/K3 avg pools (parallel)
//   pack kernels: weights -> WMMA B-fragment layout f16 (zero padded)
//   im2col kernels: activations -> padded [M,Kpad] f16 A panels
//   unified WMMA GEMM (conv2 / conv3 / fc1 / fc2 / fc3):
//       B panel staged to LDS via Tensor Data Mover once per block,
//       4 M-tiles per wave (register blocking): per k-step
//       2x ds_load_b128 (B) + 8x global_load_b128 (A) + 4x v_wmma
//   iaf scan kernels between layers (sequential in T per neuron, L2-resident)
// ---------------------------------------------------------------------------

typedef _Float16 f16;
typedef _Float16 v16h __attribute__((ext_vector_type(16)));
typedef _Float16 v8h  __attribute__((ext_vector_type(8)));
typedef float    v8f  __attribute__((ext_vector_type(8)));
typedef unsigned int v4u __attribute__((ext_vector_type(4)));
typedef int          v4i __attribute__((ext_vector_type(4)));
typedef int          v8i __attribute__((ext_vector_type(8)));

#define T_STEPS 4096

__device__ __forceinline__ v8f wmma_step(v16h a, v16h b, v8f c) {
    return __builtin_amdgcn_wmma_f32_16x16x32_f16(
        false, a, false, b, (short)0, c, false, false);
}

__device__ __forceinline__ float iaf_step(float& v, float pre) {
    v += pre;
    float s = floorf(v);
    s = s > 0.f ? s : 0.f;
    v -= s;
    return s;
}

// ---- stage a contiguous f16 panel into LDS via the Tensor Data Mover -------
__device__ __forceinline__ void stage_b(f16* lds, const f16* src, int elems) {
#if __has_builtin(__builtin_amdgcn_tensor_load_to_lds)
    if (threadIdx.x < 32) {   // one wave issues the DMA (EXEC ignored by TDM)
        unsigned lds_addr = (unsigned)(unsigned long long)(size_t)lds; // low 32b = LDS offset
        unsigned long long ga = (unsigned long long)(size_t)src;
        unsigned ne = (unsigned)elems;
        // D# group 0: count=1 | lds_addr | global_addr[56:0] | type=2
        v4u g0;
        g0[0] = 1u;
        g0[1] = lds_addr;
        g0[2] = (unsigned)(ga & 0xffffffffull);
        g0[3] = (unsigned)((ga >> 32) & 0x01ffffffull) | (2u << 30);
        // D# group 1: data_size=1(2B); tensor_dim0=ne, tensor_dim1=1,
        // tile_dim0=ne, tile_dim1=1, tensor_dim0_stride=ne (single-row tile)
        v8i g1;
        g1[0] = (int)(1u << 16);                                   // data_size=2B
        g1[1] = (int)((ne & 0xffffu) << 16);                       // dim0[15:0]
        g1[2] = (int)(((ne >> 16) & 0xffffu) | (1u << 16));        // dim0[31:16], dim1=1
        g1[3] = (int)((ne & 0xffffu) << 16);                       // tile_dim0=ne
        g1[4] = (int)(1u);                                         // tile_dim1=1
        g1[5] = (int)ne;                                           // stride0[31:0]
        g1[6] = 0;
        g1[7] = 0;
        v4i g2 = {0, 0, 0, 0};
        v4i g3 = {0, 0, 0, 0};
#if defined(__clang_major__) && (__clang_major__ >= 23)
        v8i g4 = {0, 0, 0, 0, 0, 0, 0, 0};
        __builtin_amdgcn_tensor_load_to_lds(g0, g1, g2, g3, g4, 0);
#else
        __builtin_amdgcn_tensor_load_to_lds(g0, g1, g2, g3, 0);
#endif
        __builtin_amdgcn_s_wait_tensorcnt(0);
    }
#else
    for (int i = threadIdx.x; i < elems; i += blockDim.x) lds[i] = src[i];
#endif
    __syncthreads();
}

// ---------------- K1: conv1 (2ch,2x2) fused with IAF scan -------------------
// x:[T,2,34,34] f32, w1:[10,2,2,2] -> c1 spikes f16 [T,10,33,33]
__global__ void k_conv1_iaf(const float* __restrict__ x,
                            const float* __restrict__ w1,
                            f16* __restrict__ c1) {
    int n = blockIdx.x * blockDim.x + threadIdx.x;
    if (n >= 10890) return;
    int c = n / 1089, rem = n % 1089, y = rem / 33, xx = rem % 33;
    float w[8];
#pragma unroll
    for (int i = 0; i < 8; ++i) {
        int ci = i >> 2, ky = (i >> 1) & 1, kx = i & 1;
        w[i] = w1[((c * 2 + ci) * 2 + ky) * 2 + kx];
    }
    float v = 0.f;
    for (int t = 0; t < T_STEPS; ++t) {
        const float* xt = x + (size_t)t * (2 * 34 * 34);
        float pre = 0.f;
#pragma unroll
        for (int i = 0; i < 8; ++i) {
            int ci = i >> 2, ky = (i >> 1) & 1, kx = i & 1;
            pre += w[i] * xt[(ci * 34 + (y + ky)) * 34 + (xx + kx)];
        }
        c1[(size_t)t * 10890 + n] = (f16)iaf_step(v, pre);
    }
}

// ---------------- pools -----------------------------------------------------
__global__ void k_pool3(const f16* __restrict__ c1, f16* __restrict__ p1) {
    int idx = blockIdx.x * blockDim.x + threadIdx.x;
    if (idx >= T_STEPS * 1210) return;
    int t = idx / 1210, r = idx % 1210;
    int c = r / 121, q = r % 121, yy = q / 11, xx = q % 11;
    const f16* base = c1 + (size_t)t * 10890 + c * 1089 + (yy * 3) * 33 + xx * 3;
    float s = 0.f;
#pragma unroll
    for (int j = 0; j < 3; ++j)
#pragma unroll
        for (int i = 0; i < 3; ++i) s += (float)base[j * 33 + i];
    p1[idx] = (f16)(s * (1.f / 9.f));
}

__global__ void k_pool4(const f16* __restrict__ c1, float* __restrict__ p1a) {
    int idx = blockIdx.x * blockDim.x + threadIdx.x;
    if (idx >= T_STEPS * 640) return;
    int t = idx / 640, r = idx % 640;
    int c = r / 64, q = r % 64, yy = q / 8, xx = q % 8;
    const f16* base = c1 + (size_t)t * 10890 + c * 1089 + (yy * 4) * 33 + xx * 4;
    float s = 0.f;
#pragma unroll
    for (int j = 0; j < 4; ++j)
#pragma unroll
        for (int i = 0; i < 4; ++i) s += (float)base[j * 33 + i];
    p1a[idx] = s * (1.f / 16.f);
}

// ---------------- weight pack: W f32 [Nact,K] -> B fragments f16 ------------
// pack layout: [((tn*ksteps + kb)*32 + lane)*16 + i], zero padded
__global__ void k_pack_b(const float* __restrict__ W, int Nact, int K,
                         int ksteps, int total, f16* __restrict__ pack) {
    int idx = blockIdx.x * blockDim.x + threadIdx.x;
    if (idx >= total) return;
    int i    = idx & 15;
    int lane = (idx >> 4) & 31;
    int kb   = (idx >> 9) % ksteps;
    int tn   = idx / (ksteps * 512);
    int n = tn * 16 + (lane & 15);
    int k = kb * 32 + (i & 7) + ((lane & 16) ? 8 : 0) + ((i & 8) ? 16 : 0);
    f16 v = (f16)0.f;
    if (n < Nact && k < K) v = (f16)W[(size_t)n * K + k];
    pack[idx] = v;
}

// ---------------- im2col for conv2: p1 [T,10,11,11] -> A [T*64, 160] --------
__global__ void k_im2col2(const f16* __restrict__ p1, f16* __restrict__ a2) {
    int idx = blockIdx.x * blockDim.x + threadIdx.x;
    if (idx >= T_STEPS * 64 * 160) return;
    int m = idx / 160, k = idx % 160;
    int t = m >> 6, oy = (m >> 3) & 7, ox = m & 7;
    int ci = k >> 4, rr = (k >> 2) & 3, ss = k & 3;
    a2[idx] = p1[(((size_t)t * 10 + ci) * 11 + (oy + rr)) * 11 + (ox + ss)];
}

// ---------------- im2col for conv3: m_h [T,10,8,8] -> A [T*49, 64] ----------
__global__ void k_im2col3(const f16* __restrict__ m_h, f16* __restrict__ a3) {
    int idx = blockIdx.x * blockDim.x + threadIdx.x;
    if (idx >= T_STEPS * 49 * 64) return;
    int m = idx >> 6, k = idx & 63;
    int t = m / 49, pp = m % 49, oy = pp / 7, ox = pp % 7;
    f16 v = (f16)0.f;
    if (k < 40) {
        int ci = k >> 2, rr = (k >> 1) & 1, ss = k & 1;
        v = m_h[(((size_t)t * 10 + ci) * 8 + (oy + rr)) * 8 + (ox + ss)];
    }
    a3[idx] = v;
}

// ---------------- unified WMMA GEMM: out[M,ldo] = A[M,Kpad] * packB^T -------
// grid: (tilesM/(8*4), tilesN), block 256 = 8 waves, 4 M-tiles per wave.
// Per k-step: 2 ds_load_b128 (B, shared) + 4 x (2 global_load_b128 + wmma).
__global__ void k_wmma_gemm(const f16* __restrict__ A, int Kpad,
                            const f16* __restrict__ pack, int ksteps,
                            int Nact, float* __restrict__ outp, int ldo) {
    __shared__ __align__(64) f16 lds_b[5 * 512];   // up to 5 k-steps
    int tn = blockIdx.y;
    stage_b(lds_b, pack + (size_t)tn * ksteps * 512, ksteps * 512);

    int lane = threadIdx.x & 31;
    int tm0  = (blockIdx.x * (blockDim.x >> 5) + (threadIdx.x >> 5)) * 4;
    const f16* arow = A + (size_t)(tm0 * 16 + (lane & 15)) * Kpad
                        + ((lane & 16) ? 8 : 0);
    const size_t tile_stride = (size_t)16 * Kpad;   // A rows per M-tile

    v8f acc0 = {}, acc1 = {}, acc2 = {}, acc3 = {};
    for (int kb = 0; kb < ksteps; ++kb) {
        v16h b = *(const v16h*)(&lds_b[(kb * 32 + lane) * 16]);
        const f16* ap = arow + kb * 32;
#pragma unroll
        for (int q = 0; q < 4; ++q) {
            const f16* aq = ap + q * tile_stride;
            v8h lo = *(const v8h*)aq;          // K slots 0..7   (+koff)
            v8h hi = *(const v8h*)(aq + 16);   // K slots 16..23 (+koff)
            v16h a = __builtin_shufflevector(lo, hi, 0, 1, 2, 3, 4, 5, 6, 7,
                                             8, 9, 10, 11, 12, 13, 14, 15);
            if      (q == 0) acc0 = wmma_step(a, b, acc0);
            else if (q == 1) acc1 = wmma_step(a, b, acc1);
            else if (q == 2) acc2 = wmma_step(a, b, acc2);
            else             acc3 = wmma_step(a, b, acc3);
        }
    }

    int nn = tn * 16 + (lane & 15);
    if (nn < Nact) {
        int moff = ((lane & 16) ? 8 : 0);
#pragma unroll
        for (int q = 0; q < 4; ++q) {
            v8f acc = (q == 0) ? acc0 : (q == 1) ? acc1 : (q == 2) ? acc2 : acc3;
#pragma unroll
            for (int r = 0; r < 8; ++r) {
                int mm = (tm0 + q) * 16 + r + moff;
                outp[(size_t)mm * ldo + nn] = acc[r];
            }
        }
    }
}

// ---------------- iaf(c2) + merge: pre2m [T*64,10] + p1a -> m_h [T,10,8,8] --
__global__ void k_scan_merge(const float* __restrict__ pre2m,
                             const float* __restrict__ p1a,
                             f16* __restrict__ m_h) {
    int n = blockIdx.x * blockDim.x + threadIdx.x;   // n = c*64 + oy*8+ox
    if (n >= 640) return;
    int c = n / 64, sp = n % 64;
    float v = 0.f;
    for (int t = 0; t < T_STEPS; ++t) {
        float s = iaf_step(v, pre2m[(size_t)t * 640 + sp * 10 + c]);
        size_t idx = (size_t)t * 640 + n;
        m_h[idx] = (f16)(p1a[idx] + s);
    }
}

// ---------------- generic iaf scan -> padded f16 (and/or f32) outputs -------
__global__ void k_scan_pad(const float* __restrict__ pre, int ldsrc, int N,
                           int Kpad, f16* __restrict__ outh,
                           float* __restrict__ outf, int ldf) {
    int n = blockIdx.x * blockDim.x + threadIdx.x;
    if (n >= Kpad) return;
    if (n >= N) {                      // zero-fill K padding columns
        for (int t = 0; t < T_STEPS; ++t) outh[(size_t)t * Kpad + n] = (f16)0.f;
        return;
    }
    float v = 0.f;
    for (int t = 0; t < T_STEPS; ++t) {
        if (t + 8 < T_STEPS) __builtin_prefetch(pre + (size_t)(t + 8) * ldsrc + n, 0, 1);
        float s = iaf_step(v, pre[(size_t)t * ldsrc + n]);
        if (outh) outh[(size_t)t * Kpad + n] = (f16)s;
        if (outf) outf[(size_t)t * ldf + n] = s;
    }
}

// ---------------------------------------------------------------------------
extern "C" void kernel_launch(void* const* d_in, const int* in_sizes, int n_in,
                              void* d_out, int out_size, void* d_ws, size_t ws_size,
                              hipStream_t stream) {
    const float* x   = (const float*)d_in[0];
    const float* w1  = (const float*)d_in[1];
    const float* w2  = (const float*)d_in[2];   // [10,10,4,4] == [10,160]
    const float* w3  = (const float*)d_in[3];   // [1,10,2,2]  == [1,40]
    const float* fw1 = (const float*)d_in[4];   // [100,49]
    const float* fw2 = (const float*)d_in[5];   // [100,100]
    const float* fw3 = (const float*)d_in[6];   // [10,100]
    float* out = (float*)d_out;
    (void)in_sizes; (void)n_in; (void)out_size; (void)ws_size;

    char* p = (char*)d_ws;
    auto alloc = [&](size_t bytes) -> char* {
        char* r = p;
        p += (bytes + 255) & ~(size_t)255;
        return r;
    };
    const size_t T = T_STEPS;
    f16*   c1    = (f16*)  alloc(T * 10890 * 2);       // 89.2 MB
    f16*   p1    = (f16*)  alloc(T * 1210 * 2);        //  9.9 MB
    float* p1a   = (float*)alloc(T * 640 * 4);         // 10.5 MB
    f16*   a2    = (f16*)  alloc(T * 64 * 160 * 2);    // 83.9 MB im2col conv2
    float* pre2m = (float*)alloc(T * 64 * 10 * 4);     // 10.5 MB
    f16*   m_h   = (f16*)  alloc(T * 640 * 2);         //  5.2 MB
    f16*   a3    = (f16*)  alloc(T * 49 * 64 * 2);     // 25.7 MB im2col conv3
    float* pre3  = (float*)alloc(T * 49 * 4);
    f16*   c3p   = (f16*)  alloc(T * 64 * 2);          // padded [T,64]
    float* pref1 = (float*)alloc(T * 100 * 4);
    f16*   f1p   = (f16*)  alloc(T * 128 * 2);         // padded [T,128]
    float* pref2 = (float*)alloc(T * 100 * 4);
    f16*   f2p   = (f16*)  alloc(T * 128 * 2);
    float* pref3 = (float*)alloc(T * 10 * 4);
    f16* pk2  = (f16*)alloc(1 * 5 * 512 * 2);          // conv2 pack (tilesN=1)
    f16* pk3  = (f16*)alloc(1 * 2 * 512 * 2);          // conv3 pack
    f16* pkf1 = (f16*)alloc(7 * 2 * 512 * 2);          // fc1 pack
    f16* pkf2 = (f16*)alloc(7 * 4 * 512 * 2);          // fc2 pack
    f16* pkf3 = (f16*)alloc(1 * 4 * 512 * 2);          // fc3 pack

    // weight packs (independent of activations -> issue first)
    k_pack_b<<<(2560 + 255) / 256, 256, 0, stream>>>(w2, 10, 160, 5, 2560, pk2);
    k_pack_b<<<(1024 + 255) / 256, 256, 0, stream>>>(w3, 1, 40, 2, 1024, pk3);
    k_pack_b<<<(7168 + 255) / 256, 256, 0, stream>>>(fw1, 100, 49, 2, 7168, pkf1);
    k_pack_b<<<(14336 + 255) / 256, 256, 0, stream>>>(fw2, 100, 100, 4, 14336, pkf2);
    k_pack_b<<<(2048 + 255) / 256, 256, 0, stream>>>(fw3, 10, 100, 4, 2048, pkf3);

    // conv1 + iaf
    k_conv1_iaf<<<(10890 + 255) / 256, 256, 0, stream>>>(x, w1, c1);
    // pools
    k_pool3<<<(int)((T * 1210 + 255) / 256), 256, 0, stream>>>(c1, p1);
    k_pool4<<<(int)((T * 640 + 255) / 256), 256, 0, stream>>>(c1, p1a);
    // conv2: im2col + WMMA GEMM (16384 M-tiles / 4 per wave / 8 waves = 512)
    k_im2col2<<<(int)((T * 64 * 160 + 255) / 256), 256, 0, stream>>>(p1, a2);
    k_wmma_gemm<<<dim3(512, 1), 256, 0, stream>>>(a2, 160, pk2, 5, 10, pre2m, 10);
    // iaf(c2) + merge
    k_scan_merge<<<3, 256, 0, stream>>>(pre2m, p1a, m_h);
    // conv3: im2col + WMMA GEMM (12544 M-tiles -> 392 blocks)
    k_im2col3<<<(int)((T * 49 * 64 + 255) / 256), 256, 0, stream>>>(m_h, a3);
    k_wmma_gemm<<<dim3(392, 1), 256, 0, stream>>>(a3, 64, pk3, 2, 1, pre3, 1);
    // iaf(c3) -> padded [T,64]
    k_scan_pad<<<1, 64, 0, stream>>>(pre3, 49, 49, 64, c3p, nullptr, 0);
    // fc1 (256 M-tiles -> 8 blocks x 7 N-tiles)
    k_wmma_gemm<<<dim3(8, 7), 256, 0, stream>>>(c3p, 64, pkf1, 2, 100, pref1, 100);
    k_scan_pad<<<1, 128, 0, stream>>>(pref1, 100, 100, 128, f1p, nullptr, 0);
    // fc2
    k_wmma_gemm<<<dim3(8, 7), 256, 0, stream>>>(f1p, 128, pkf2, 4, 100, pref2, 100);
    k_scan_pad<<<1, 128, 0, stream>>>(pref2, 100, 100, 128, f2p, nullptr, 0);
    // fc3
    k_wmma_gemm<<<dim3(8, 1), 256, 0, stream>>>(f2p, 128, pkf3, 4, 10, pref3, 10);
    // final iaf -> d_out [T,10] f32
    k_scan_pad<<<1, 32, 0, stream>>>(pref3, 10, 10, 10, nullptr, out, 10);
}
